// DenseEdgeModel_18056042512642
// MI455X (gfx1250) — matrix operand order repaired and checked by reading, not compiled, not run on hardware
//
#include <hip/hip_runtime.h>
#include <hip/hip_bf16.h>

// ---------------------------------------------------------------------------
// DenseEdgeModel fused implementation for gfx1250 (MI455X).
//
// ~35 GFLOP total, 97% in the two 256x256 GEMMs over the pairwise (xp-xc)^2
// tensor (134MB fp32 if materialized -- never materialized here).  Each
// workgroup builds a [128 pairs x 256 ch] bf16 tile in LDS and runs both
// GEMMs tile-resident with v_wmma_f32_16x16x32_bf16 (f32 accumulate).
// A-fragments are register-hoisted (one LDS read per layer), and two o-tiles
// are processed concurrently to break the WMMA RAW chain.
// Pre stack (0.8 GFLOP, precision-critical) uses v_wmma_f32_16x16x4_f32.
// ---------------------------------------------------------------------------

typedef __attribute__((ext_vector_type(2)))  float  v2f;
typedef __attribute__((ext_vector_type(4)))  float  v4f;
typedef __attribute__((ext_vector_type(8)))  float  v8f;
typedef __attribute__((ext_vector_type(16))) __bf16 v16bf;
typedef __attribute__((ext_vector_type(8)))  __bf16 v8bf;
typedef __attribute__((ext_vector_type(4)))  __bf16 v4bf;

#define CCH   256
#define LDSTR 264    // LDS row stride in bf16 halves (528B = 33*16: 16B aligned, bank pad)
#define MROWS 128    // pairs per workgroup tile (16 p x 8 q)

#if defined(__has_builtin)
#if __has_builtin(__builtin_amdgcn_wmma_f32_16x16x4_f32)
#define HAVE_WMMA_F32 1
#endif
#endif

// ---- transpose x[b][c][hw] -> xt[node][c], node = b*1024 + hw -------------
__global__ void k_xpose(const float* __restrict__ x, float* __restrict__ xt) {
  int i  = blockIdx.x * 256 + threadIdx.x;      // coalesced read index
  int hw = i & 1023;
  int bc = i >> 10;
  int c  = bc & 255;
  int b  = bc >> 8;
  xt[(((b << 10) + hw) << 8) + c] = x[i];
}

// ---- pre layer: out[2048][256] = act(in[2048][256] @ W[o][c]^T + bias) ----
__global__ void k_pre_gemm(const float* __restrict__ in, const float* __restrict__ w,
                           const float* __restrict__ bias, float* __restrict__ out,
                           int relu) {
#if HAVE_WMMA_F32
  int wid  = threadIdx.x >> 5;
  int lane = threadIdx.x & 31;
  int half = lane >> 4;
  int lr   = lane & 15;
  int tile = blockIdx.x * 8 + wid;              // 2048 tiles = 128 mtiles x 16 otiles
  int mt   = tile >> 4;
  int ot   = tile & 15;
  int m    = mt * 16 + lr;                      // A row for this lane
  int o    = ot * 16 + lr;                      // B column for this lane
  v8f acc = {};
  const float* arow = in + m * CCH;
  const float* brow = w  + o * CCH;
#pragma unroll 8
  for (int kk = 0; kk < 64; ++kk) {
    int c = kk * 4 + half * 2;                  // 32-bit A 16x4 / B 4x16 lane layout
    v2f a = *(const v2f*)(arow + c);
    v2f b = *(const v2f*)(brow + c);
    acc = __builtin_amdgcn_wmma_f32_16x16x4_f32(false, a, false, b, (short)0, acc,
                                                false, false);
  }
  float bv = bias[o];
#pragma unroll
  for (int r = 0; r < 8; ++r) {
    int row = mt * 16 + r + 8 * half;           // f32 C/D layout: M = r + 8*(lane/16)
    float f = acc[r] + bv;
    if (relu) f = fmaxf(f, 0.0f);
    out[row * CCH + o] = f;
  }
#else
  for (int rr = 0; rr < 8; ++rr) {
    int lin = (blockIdx.x * 256 + threadIdx.x) * 8 + rr;
    int m = lin >> 8, o = lin & 255;
    float s = bias[o];
    for (int c = 0; c < CCH; ++c) s += in[m * CCH + c] * w[o * CCH + c];
    if (relu) s = fmaxf(s, 0.0f);
    out[m * CCH + o] = s;
  }
#endif
}

// ---- gather node rows by global flat index --------------------------------
__global__ void k_gather(const float* __restrict__ feat, const int* __restrict__ pidx,
                         const int* __restrict__ cidx, float* __restrict__ xp,
                         float* __restrict__ xc) {
  int row = blockIdx.x;                         // 0..511 primary, 512..1023 compare
  int c   = threadIdx.x;
  if (row < 512) {
    int src = pidx[row];
    xp[row * CCH + c] = feat[src * CCH + c];
  } else {
    int r   = row - 512;
    int src = cidx[r];
    xc[r * CCH + c] = feat[src * CCH + c];
  }
}

// ---- convert post weights [2][256][256] f32 -> bf16 -----------------------
__global__ void k_wcvt(const float* __restrict__ w, __bf16* __restrict__ wbf) {
  int i = blockIdx.x * 256 + threadIdx.x;
  wbf[i] = (__bf16)w[i];
}

static __device__ __forceinline__ v16bf cat16(v8bf lo, v8bf hi) {
  return __builtin_shufflevector(lo, hi, 0, 1, 2, 3, 4, 5, 6, 7, 8, 9, 10, 11, 12, 13,
                                 14, 15);
}

static __device__ __forceinline__ v8f wmma_bf16(v16bf a, v16bf b, v8f c) {
  return __builtin_amdgcn_wmma_f32_16x16x32_bf16(false, a, false, b, (short)0, c,
                                                 false, false);
}

// One 256->256 GEMM layer, tile resident.  A in LDS ([m][c] bf16, LDSTR rows),
// hoisted into registers once (8 x v16bf = 64 VGPRs); B = W row-major bf16 in
// global (L2-hot).  Two o-tiles in flight to break the WMMA RAW chain.
// D -> Dst LDS [m][o] bf16 after bias+ReLU.  Wave owns m-rows [wid*16, +16).
static __device__ __forceinline__ void gemm_layer(const __bf16* A, const __bf16* W,
                                                  const float* bias, __bf16* Dst,
                                                  int wid, int half, int lr) {
  const __bf16* arow = A + (wid * 16 + lr) * LDSTR;   // 16-bit A layout: row = lane%16
  v16bf av[8];
#pragma unroll
  for (int kk = 0; kk < 8; ++kk) {                    // hoist A fragment: 16 ds loads
    int ca = kk * 32 + half * 8;                      // K = (lane/16)*8..+8, +16..+24
    v8bf alo = *(const v8bf*)(arow + ca);
    v8bf ahi = *(const v8bf*)(arow + ca + 16);
    av[kk] = cat16(alo, ahi);
  }
  for (int op = 0; op < 8; ++op) {                    // 8 pairs of o-tiles
    int o0 = op * 32 + lr;
    int o1 = o0 + 16;
    const __bf16* w0 = W + o0 * CCH;
    const __bf16* w1 = W + o1 * CCH;
    v8f acc0 = {}, acc1 = {};
#pragma unroll
    for (int kk = 0; kk < 8; ++kk) {
      int cb = kk * 32 + half * 8;                    // B: K = (lane/16)*8..+8, +16..+24
      v8bf b0lo = *(const v8bf*)(w0 + cb);
      v8bf b0hi = *(const v8bf*)(w0 + cb + 16);
      v8bf b1lo = *(const v8bf*)(w1 + cb);
      v8bf b1hi = *(const v8bf*)(w1 + cb + 16);
      acc0 = wmma_bf16(av[kk], cat16(b0lo, b0hi), acc0);   // two independent
      acc1 = wmma_bf16(av[kk], cat16(b1lo, b1hi), acc1);   // WMMA chains
    }
    float bv0 = bias[o0];
    float bv1 = bias[o1];
#pragma unroll
    for (int r = 0; r < 8; ++r) {
      int m = wid * 16 + r + 8 * half;                // f32 C/D layout
      float f0 = fmaxf(acc0[r] + bv0, 0.0f);          // both post layers have ReLU
      float f1 = fmaxf(acc1[r] + bv1, 0.0f);
      Dst[m * LDSTR + o0] = (__bf16)f0;
      Dst[m * LDSTR + o1] = (__bf16)f1;
    }
  }
}

// ---- fused post stack: xx tile -> GEMM1 -> GEMM2 -> 256->2 projection -----
__global__ void k_post(const float* __restrict__ xp, const float* __restrict__ xc,
                       const __bf16* __restrict__ w1, const __bf16* __restrict__ w2,
                       const float* __restrict__ pb, const float* __restrict__ w3,
                       const float* __restrict__ b3, float* __restrict__ out) {
  extern __shared__ __bf16 lds[];
  __bf16* bufX = lds;                  // xx tile, later reused for g2
  __bf16* bufG = lds + MROWS * LDSTR;  // g1 tile
  int bid = blockIdx.x;
  int b   = bid >> 9;
  int rem = bid & 511;
  int p0  = (rem >> 5) * 16;
  int q0  = (rem & 31) * 8;
  int t   = threadIdx.x;

  // stage 1: bufX[m][c] = bf16((xp[p][c]-xc[q][c])^2), m = plocal*8 + qlocal
  {
    int m  = t >> 1;
    int ch = (t & 1) * 128;
    int p  = p0 + (m >> 3);
    int q  = q0 + (m & 7);
    const float* xprow = xp + (b * 256 + p) * CCH + ch;
    const float* xcrow = xc + (b * 256 + q) * CCH + ch;
    __bf16* dst = bufX + m * LDSTR + ch;
#pragma unroll
    for (int c = 0; c < 128; c += 4) {
      v4f a = *(const v4f*)(xprow + c);
      v4f d = *(const v4f*)(xcrow + c);
      v4f e = a - d;
      v4f s = e * e;
      v4bf r;
      r[0] = (__bf16)s[0]; r[1] = (__bf16)s[1];
      r[2] = (__bf16)s[2]; r[3] = (__bf16)s[3];
      *(v4bf*)(dst + c) = r;
    }
  }
  __syncthreads();

  int wid  = t >> 5;
  int lane = t & 31;
  int half = lane >> 4;
  int lr   = lane & 15;

  gemm_layer(bufX, w1, pb, bufG, wid, half, lr);        // layer 1 (xx -> g1)
  __syncthreads();
  gemm_layer(bufG, w2, pb + 256, bufX, wid, half, lr);  // layer 2 (g1 -> g2)
  __syncthreads();

  // stage 4: out[b][j][p][q] = sum_o g2[m][o] * w3[j][o] + b3[j]
  {
    int j = t & 1;
    int m = t >> 1;
    const __bf16* g = bufX + m * LDSTR;
    const float* wr = w3 + j * CCH;
    float s = b3[j];
    for (int o = 0; o < CCH; o += 8) {
      v8bf g8 = *(const v8bf*)(g + o);
#pragma unroll
      for (int u = 0; u < 8; ++u) s += (float)g8[u] * wr[o + u];
    }
    int p = p0 + (m >> 3);
    int q = q0 + (m & 7);
    out[((b * 2 + j) * 256 + p) * 256 + q] = s;
  }
}

extern "C" void kernel_launch(void* const* d_in, const int* in_sizes, int n_in,
                              void* d_out, int out_size, void* d_ws, size_t ws_size,
                              hipStream_t stream) {
  (void)in_sizes; (void)n_in; (void)out_size; (void)ws_size;
  const float* x      = (const float*)d_in[0];
  const int*   pidx   = (const int*)d_in[1];
  const int*   cidx   = (const int*)d_in[2];
  const float* pre_w  = (const float*)d_in[3];   // [3][256][256]
  const float* pre_b  = (const float*)d_in[4];   // [3][256]
  const float* post_w = (const float*)d_in[5];   // [2][256][256]
  const float* post_b = (const float*)d_in[6];   // [2][256]
  const float* pow_w  = (const float*)d_in[7];   // [2][256]
  const float* pow_b  = (const float*)d_in[8];   // [2]
  float* out = (float*)d_out;

  char*   ws   = (char*)d_ws;
  float*  buf0 = (float*)(ws);                           // 2 MB
  float*  buf1 = (float*)(ws + (2u << 20));              // 2 MB
  float*  xpT  = (float*)(ws + (4u << 20));              // 512 KB
  float*  xcT  = (float*)(ws + (4u << 20) + (512u << 10));
  __bf16* wbf  = (__bf16*)(ws + (5u << 20));             // 256 KB (w1,w2 bf16)

  k_xpose<<<2048, 256, 0, stream>>>(x, buf0);
  k_pre_gemm<<<256, 256, 0, stream>>>(buf0, pre_w,          pre_b,       buf1, 1);
  k_pre_gemm<<<256, 256, 0, stream>>>(buf1, pre_w + 65536,  pre_b + 256, buf0, 1);
  k_pre_gemm<<<256, 256, 0, stream>>>(buf0, pre_w + 131072, pre_b + 512, buf1, 0);
  k_gather<<<1024, 256, 0, stream>>>(buf1, pidx, cidx, xpT, xcT);
  k_wcvt<<<512, 256, 0, stream>>>(post_w, wbf);

  size_t shmem = (size_t)2 * MROWS * LDSTR * sizeof(__bf16);  // 132 KB
  k_post<<<1024, 256, shmem, stream>>>(xpT, xcT, wbf, wbf + 65536, post_b, pow_w,
                                       pow_b, out);
}
